// GATv2Encoder_33285996544639
// MI455X (gfx1250) — matrix operand order repaired and checked by reading, not compiled, not run on hardware
//
#include <hip/hip_runtime.h>
#include <hip/hip_bf16.h>

// GATv2 encoder for gfx1250 (MI455X). bf16 WMMA for all GEMMs; fused
// edge-feature GEMM + attention-logit epilogue so the [E,512] ee tensor is
// never materialized. B matrices pre-packed to fragment-major bf16 so the
// WMMA inner loop is pure b128 loads + v_wmma. Workspace ~108 MB.

#define NN    20000
#define EEDG  160000
#define DINF  384
#define HIDF  128
#define NHEAD 4
#define NLAY  2
#define NEGS  0.2f
#define EATOT (EEDG + NN)

typedef __attribute__((ext_vector_type(16))) __bf16   v16bf;
typedef __attribute__((ext_vector_type(8)))  float    v8f;
typedef __attribute__((ext_vector_type(8)))  unsigned v8u;

// f32 pair -> packed bf16 dword (RNE). Single v_cvt_pk_bf16_f32 if the
// toolchain has it; otherwise RNE adds + one v_perm_b32 merge.
__device__ __forceinline__ unsigned pk_bf16(float a, float b) {
#if __has_builtin(__builtin_amdgcn_cvt_pk_bf16_f32)
  auto t = __builtin_amdgcn_cvt_pk_bf16_f32(a, b);
  return __builtin_bit_cast(unsigned, t);
#else
  unsigned ua = __float_as_uint(a), ub = __float_as_uint(b);
  ua += 0x7FFFu + ((ua >> 16) & 1u);
  ub += 0x7FFFu + ((ub >> 16) & 1u);
  // result bytes {ua.b2, ua.b3, ub.b2, ub.b3} = (ua>>16)|(ub&0xFFFF0000)
  return __builtin_amdgcn_perm(ub, ua, 0x07060302u);
#endif
}
__device__ __forceinline__ unsigned short f2bfu(float f) {
  unsigned u = __float_as_uint(f);
  u += 0x7FFFu + ((u >> 16) & 1u);
  return (unsigned short)(u >> 16);
}

// Monotonic float<->uint key for atomicMax-based segment max.
__device__ __forceinline__ unsigned fkey(float f) {
  unsigned u = __float_as_uint(f);
  return (u >> 31) ? ~u : (u | 0x80000000u);
}
__device__ __forceinline__ float fval(unsigned k) {
  unsigned u = (k & 0x80000000u) ? (k & 0x7FFFFFFFu) : ~k;
  return __uint_as_float(u);
}

// ---- A fragment (16x32 bf16, ISA 7.12.2): lanes 0-15/16-31 both rows 0-15;
// lo lanes K={0..7,16..23}, hi lanes K={8..15,24..31}. Vectorized b128 loads,
// predicate hoisted so no per-element exec toggling.
__device__ __forceinline__ v16bf pack_a(const float4& f0, const float4& f1,
                                        const float4& f2, const float4& f3) {
  v8u u;
  u[0] = pk_bf16(f0.x, f0.y); u[1] = pk_bf16(f0.z, f0.w);
  u[2] = pk_bf16(f1.x, f1.y); u[3] = pk_bf16(f1.z, f1.w);
  u[4] = pk_bf16(f2.x, f2.y); u[5] = pk_bf16(f2.z, f2.w);
  u[6] = pk_bf16(f3.x, f3.y); u[7] = pk_bf16(f3.z, f3.w);
  return __builtin_bit_cast(v16bf, u);
}
__device__ __forceinline__ v16bf load_a_frag(const float* A, int lda,
                                             int row0, int k0, int mMax) {
  int lane = threadIdx.x & 31;
  int r = row0 + (lane & 15);
  const float* Ar = A + (size_t)r * lda + k0 + (lane >> 4) * 8;
  float4 f0, f1, f2, f3;
  if (r < mMax) {
    f0 = *(const float4*)(Ar);
    f1 = *(const float4*)(Ar + 4);
    f2 = *(const float4*)(Ar + 16);
    f3 = *(const float4*)(Ar + 20);
  } else {
    f0 = f1 = f2 = f3 = make_float4(0.f, 0.f, 0.f, 0.f);
  }
  return pack_a(f0, f1, f2, f3);
}
__device__ __forceinline__ v16bf load_a_frag_nc(const float* A, int lda,
                                                int row0, int k0) {
  int lane = threadIdx.x & 31;
  int r = row0 + (lane & 15);
  const float* Ar = A + (size_t)r * lda + k0 + (lane >> 4) * 8;
  float4 f0 = *(const float4*)(Ar);
  float4 f1 = *(const float4*)(Ar + 4);
  float4 f2 = *(const float4*)(Ar + 16);
  float4 f3 = *(const float4*)(Ar + 20);
  if (k0 + 32 < DINF) __builtin_prefetch(Ar + 32, 0, 1);  // global_prefetch_b8
  return pack_a(f0, f1, f2, f3);
}

// ---- B fragment from fragment-major packed bf16: lane's 16 values are
// contiguous -> exactly two global_load_b128, no conversion VALU.
__device__ __forceinline__ v16bf load_b_pk(const unsigned short* PB, int tile) {
  int lane = threadIdx.x & 31;
  const uint4* p = (const uint4*)(PB + ((size_t)tile * 32 + lane) * 16);
  uint4 a = p[0], b = p[1];
  v8u u;
  u[0] = a.x; u[1] = a.y; u[2] = a.z; u[3] = a.w;
  u[4] = b.x; u[5] = b.y; u[6] = b.z; u[7] = b.w;
  return __builtin_bit_cast(v16bf, u);
}

// Repack B[K][Nc] f32 -> fragment-major bf16: element (k,c) -> tile
// (k/32)*(Nc/16)+(c/16), lane (c%16)+16*((k%16)/8), slot k%8+8*((k%32)/16).
__global__ void k_packB(const float* __restrict__ B, unsigned short* __restrict__ PB,
                        int K, int Nc) {
  int idx = blockIdx.x * blockDim.x + threadIdx.x, st = gridDim.x * blockDim.x;
  for (; idx < K * Nc; idx += st) {
    int k = idx / Nc, c = idx - k * Nc;
    int kk = k & 31, tile = (k >> 5) * (Nc >> 4) + (c >> 4);
    int lane = (c & 15) + (((kk >> 3) & 1) << 4);
    PB[((size_t)tile * 32 + lane) * 16 + ((kk & 7) + ((kk >> 4) << 3))] = f2bfu(B[idx]);
  }
}

// ---------------- generic GEMM: C[M,Nc] = A[M,K] @ B[K,Nc] + bias ----------
// block = 256 threads = 8 waves; tile 64 rows x 128 cols; wave = 16x64.
__global__ __launch_bounds__(256)
void k_gemm_bias(const float* __restrict__ A, const unsigned short* __restrict__ PB,
                 const float* __restrict__ bias, float* __restrict__ C,
                 int M, int K, int Nc) {
  int wave = threadIdx.x >> 5, lane = threadIdx.x & 31;
  int rt = wave & 3, ct = wave >> 2;
  int row0 = blockIdx.x * 64 + rt * 16;
  int colb = blockIdx.y * 128 + ct * 64;
  int ntc = Nc >> 4;
  v8f acc[4] = {};
  for (int k0 = 0; k0 < K; k0 += 32) {
    v16bf a = load_a_frag(A, K, row0, k0, M);
    int tb = (k0 >> 5) * ntc + (colb >> 4);
#pragma unroll
    for (int t = 0; t < 4; ++t) {
      v16bf b = load_b_pk(PB, tb + t);
      acc[t] = __builtin_amdgcn_wmma_f32_16x16x32_bf16(
          false, a, false, b, (short)0, acc[t], false, false);
    }
  }
  int hi = lane >> 4, cn = lane & 15;
#pragma unroll
  for (int t = 0; t < 4; ++t) {
    int col = colb + 16 * t + cn;
    float bv = bias[col];
#pragma unroll
    for (int v = 0; v < 8; ++v) {
      int r = row0 + v + 8 * hi;
      if (r < M) C[(size_t)r * Nc + col] = acc[t][v] + bv;
    }
  }
}

// ---- fused edge GEMM + GATv2 logit epilogue (real edges only) -------------
// block = 256 = 8 waves; tile 32 edges; wave w: rows 16*(w&1), head w>>1.
__global__ __launch_bounds__(256)
void k_edge_logits(const float* __restrict__ EA, const unsigned short* __restrict__ PWe,
                   const float* __restrict__ xl, const float* __restrict__ xr,
                   const float* __restrict__ attv_g,
                   const int* __restrict__ src, const int* __restrict__ dst,
                   float* __restrict__ logits, unsigned* __restrict__ mxkey) {
  int wave = threadIdx.x >> 5, lane = threadIdx.x & 31;
  int rt = wave & 1, head = wave >> 1;
  int e0 = blockIdx.x * 32 + rt * 16;
  int colb = head * HIDF;
  v8f acc[8] = {};
  for (int k0 = 0; k0 < DINF; k0 += 32) {
    v16bf a = load_a_frag_nc(EA, DINF, e0, k0);
    int tb = (k0 >> 5) * 32 + (colb >> 4);   // Nc = 512 -> 32 col tiles
#pragma unroll
    for (int t = 0; t < 8; ++t) {
      v16bf b = load_b_pk(PWe, tb + t);
      acc[t] = __builtin_amdgcn_wmma_f32_16x16x32_bf16(
          false, a, false, b, (short)0, acc[t], false, false);
    }
  }
  int hi = lane >> 4, cn = lane & 15;
  float av[8];
#pragma unroll
  for (int t = 0; t < 8; ++t) av[t] = attv_g[colb + 16 * t + cn];
  float part[8];
#pragma unroll
  for (int v = 0; v < 8; ++v) {
    int e = e0 + v + 8 * hi;
    int s = src[e], d = dst[e];
    size_t sb = (size_t)s * (NHEAD * HIDF) + colb;
    size_t db = (size_t)d * (NHEAD * HIDF) + colb;
    float p = 0.0f;
#pragma unroll
    for (int t = 0; t < 8; ++t) {
      int c = 16 * t + cn;
      float m = acc[t][v] + xl[sb + c] + xr[db + c];
      m = (m > 0.0f) ? m : NEGS * m;
      p += m * av[t];
    }
    part[v] = p;
  }
  // reduce across the 16 lanes of each half-wave (these are the 16 columns)
#pragma unroll
  for (int msk = 1; msk <= 8; msk <<= 1)
#pragma unroll
    for (int v = 0; v < 8; ++v) part[v] += __shfl_xor(part[v], msk, 32);
  if (cn < 8) {
    int v = cn;
    int e = e0 + v + 8 * hi;
    float lg = part[v];
    logits[(size_t)e * NHEAD + head] = lg;
    atomicMax(&mxkey[(size_t)dst[e] * NHEAD + head], fkey(lg));
  }
}

// ------------------------- small / elementwise kernels ---------------------
__global__ void k_zero(float* p, int n) {
  int i = blockIdx.x * blockDim.x + threadIdx.x, st = gridDim.x * blockDim.x;
  for (; i < n; i += st) p[i] = 0.0f;
}

__global__ void k_colmean(const float* __restrict__ EA, float* __restrict__ sums) {
  int r0 = blockIdx.x * 256;
  int t = threadIdx.x;          // 128 threads, 3 cols each
  float s0 = 0.f, s1 = 0.f, s2 = 0.f;
  int rend = min(r0 + 256, EEDG);
  for (int r = r0; r < rend; ++r) {
    const float* row = EA + (size_t)r * DINF;
    s0 += row[t]; s1 += row[t + 128]; s2 += row[t + 256];
  }
  atomicAdd(&sums[t], s0);
  atomicAdd(&sums[t + 128], s1);
  atomicAdd(&sums[t + 256], s2);
}

__global__ void k_qb(const float* __restrict__ q, const float* __restrict__ Wq,
                     const float* __restrict__ bq, const float* __restrict__ bin,
                     float* __restrict__ qb) {
  int c = threadIdx.x;          // 128 threads
  float s = bin[c] + bq[c];
  for (int d = 0; d < DINF; ++d) s += q[d] * Wq[(size_t)d * HIDF + c];
  qb[c] = s;
}

__global__ void k_eWm(const float* __restrict__ esum, const float* __restrict__ We,
                      float* __restrict__ eWm) {
  int c = blockIdx.x * blockDim.x + threadIdx.x;   // 512 total
  if (c >= NHEAD * HIDF) return;
  const float inv = 1.0f / (float)EEDG;
  float s = 0.0f;
  for (int d = 0; d < DINF; ++d) s += esum[d] * We[(size_t)d * (NHEAD * HIDF) + c];
  eWm[c] = s * inv;
}

__global__ void k_self_logits(const float* __restrict__ xl, const float* __restrict__ xr,
                              const float* __restrict__ eWm, const float* __restrict__ attv,
                              float* __restrict__ logits, unsigned* __restrict__ mxkey) {
  int idx = blockIdx.x * blockDim.x + threadIdx.x;
  if (idx >= NN * NHEAD) return;
  int n = idx >> 2, h = idx & 3;
  size_t base = (size_t)n * (NHEAD * HIDF) + h * HIDF;
  float p = 0.0f;
  for (int c = 0; c < HIDF; ++c) {
    float m = xl[base + c] + xr[base + c] + eWm[h * HIDF + c];
    m = (m > 0.0f) ? m : NEGS * m;
    p += m * attv[h * HIDF + c];
  }
  logits[(size_t)(EEDG + n) * NHEAD + h] = p;
  atomicMax(&mxkey[(size_t)n * NHEAD + h], fkey(p));
}

__global__ void k_expsum(float* __restrict__ logits, const unsigned* __restrict__ mxkey,
                         float* __restrict__ den, const int* __restrict__ dst) {
  int idx = blockIdx.x * blockDim.x + threadIdx.x;
  if (idx >= EATOT * NHEAD) return;
  int e = idx >> 2, h = idx & 3;
  int d = (e < EEDG) ? dst[e] : (e - EEDG);
  float ex = __expf(logits[idx] - fval(mxkey[(size_t)d * NHEAD + h]));
  logits[idx] = ex;
  atomicAdd(&den[(size_t)d * NHEAD + h], ex);
}

// one wave per edge: alpha, avg-attn accumulation, scatter-add of
// sum_h alpha_h * xl[src,h,:] into acc[dst,:]
__global__ __launch_bounds__(256)
void k_aggregate(const float* __restrict__ ex, const float* __restrict__ den,
                 const float* __restrict__ xl,
                 const int* __restrict__ src, const int* __restrict__ dst,
                 float* __restrict__ acc, float* __restrict__ avg_att) {
  int wave = threadIdx.x >> 5, lane = threadIdx.x & 31;
  int e = blockIdx.x * 8 + wave;
  if (e >= EATOT) return;
  int s, d;
  if (e < EEDG) { s = src[e]; d = dst[e]; }
  else          { s = e - EEDG; d = s; }
  float a[NHEAD];
#pragma unroll
  for (int h = 0; h < NHEAD; ++h)
    a[h] = ex[(size_t)e * NHEAD + h] /
           (den[(size_t)d * NHEAD + h] + 1e-16f);
  if (lane == 0 && e < EEDG)
    avg_att[e] += (a[0] + a[1] + a[2] + a[3]) * (1.0f / (NHEAD * NLAY));
  size_t sb = (size_t)s * (NHEAD * HIDF);
#pragma unroll
  for (int j = 0; j < 4; ++j) {
    int c = lane + 32 * j;
    float v = 0.0f;
#pragma unroll
    for (int h = 0; h < NHEAD; ++h) v += a[h] * xl[sb + h * HIDF + c];
    atomicAdd(&acc[(size_t)d * HIDF + c], v);
  }
}

// one wave per node: head-mean + bias + layernorm + residual (+relu)
__global__ __launch_bounds__(256)
void k_finalize(const float* __restrict__ acc, float* __restrict__ h,
                const float* __restrict__ gb, const float* __restrict__ g,
                const float* __restrict__ b, int relu) {
  int wave = threadIdx.x >> 5, lane = threadIdx.x & 31;
  int n = blockIdx.x * 8 + wave;
  if (n >= NN) return;
  float v[4]; float s = 0.0f;
#pragma unroll
  for (int j = 0; j < 4; ++j) {
    int c = lane + 32 * j;
    v[j] = acc[(size_t)n * HIDF + c] * (1.0f / NHEAD) + gb[c];
    s += v[j];
  }
#pragma unroll
  for (int m = 1; m < 32; m <<= 1) s += __shfl_xor(s, m, 32);
  float mu = s * (1.0f / HIDF);
  float qv = 0.0f;
#pragma unroll
  for (int j = 0; j < 4; ++j) { float d = v[j] - mu; qv += d * d; }
#pragma unroll
  for (int m = 1; m < 32; m <<= 1) qv += __shfl_xor(qv, m, 32);
  float rs = rsqrtf(qv * (1.0f / HIDF) + 1e-5f);
#pragma unroll
  for (int j = 0; j < 4; ++j) {
    int c = lane + 32 * j;
    float o = h[(size_t)n * HIDF + c] + (v[j] - mu) * rs * g[c] + b[c];
    if (relu) o = fmaxf(o, 0.0f);
    h[(size_t)n * HIDF + c] = o;
  }
}

__global__ void k_natt_accum(const float* __restrict__ avg_att,
                             const int* __restrict__ dst,
                             float* __restrict__ nsum, float* __restrict__ ncnt) {
  int e = blockIdx.x * blockDim.x + threadIdx.x;
  if (e >= EEDG) return;
  atomicAdd(&nsum[dst[e]], avg_att[e]);
  atomicAdd(&ncnt[dst[e]], 1.0f);
}
__global__ void k_natt_div(float* __restrict__ nsum, const float* __restrict__ ncnt,
                           float* __restrict__ tot) {
  int n = blockIdx.x * blockDim.x + threadIdx.x;
  if (n >= NN) return;
  float v = nsum[n] / fmaxf(ncnt[n], 1.0f);
  nsum[n] = v;
  atomicAdd(tot, v);
}
__global__ void k_natt_out(const float* __restrict__ nsum, const float* __restrict__ tot,
                           float* __restrict__ out_natt) {
  int n = blockIdx.x * blockDim.x + threadIdx.x;
  if (n >= NN) return;
  out_natt[n] = nsum[n] / (tot[0] + 1e-8f);
}
__global__ void k_copy(const float* __restrict__ src_, float* __restrict__ dst_, int n) {
  int i = blockIdx.x * blockDim.x + threadIdx.x, st = gridDim.x * blockDim.x;
  for (; i < n; i += st) dst_[i] = src_[i];
}

extern "C" void kernel_launch(void* const* d_in, const int* in_sizes, int n_in,
                              void* d_out, int out_size, void* d_ws, size_t ws_size,
                              hipStream_t stream) {
  const float* x    = (const float*)d_in[0];
  const int*   ei   = (const int*)d_in[1];
  const float* ea   = (const float*)d_in[2];
  const float* qe   = (const float*)d_in[3];
  const float* Win  = (const float*)d_in[4];
  const float* bin  = (const float*)d_in[5];
  const float* Wq   = (const float*)d_in[6];
  const float* bq   = (const float*)d_in[7];
  const float* Wl   = (const float*)d_in[8];
  const float* bl   = (const float*)d_in[9];
  const float* Wr   = (const float*)d_in[10];
  const float* br   = (const float*)d_in[11];
  const float* We   = (const float*)d_in[12];
  const float* att  = (const float*)d_in[13];
  const float* gbias= (const float*)d_in[14];
  const float* lng  = (const float*)d_in[15];
  const float* lnb  = (const float*)d_in[16];
  const int* src = ei;
  const int* dst = ei + EEDG;
  float* out = (float*)d_out;

  // workspace layout (floats; every block is a multiple of 4 floats = 16B)
  float* W     = (float*)d_ws;
  float* h     = W;                               // N*128
  float* xl    = h    + (size_t)NN * HIDF;        // N*512
  float* xr    = xl   + (size_t)NN * NHEAD * HIDF;
  float* lgt   = xr   + (size_t)NN * NHEAD * HIDF;// (E+N)*4
  float* den   = lgt  + (size_t)EATOT * NHEAD;    // N*4
  float* acc   = den  + (size_t)NN * NHEAD;       // N*128
  float* avg   = acc  + (size_t)NN * HIDF;        // E
  float* esum  = avg  + (size_t)EEDG;             // 384
  float* qb    = esum + DINF;                     // 128
  float* eWm   = qb   + HIDF;                     // 512
  float* nsum  = eWm  + NHEAD * HIDF;             // N
  float* ncnt  = nsum + NN;                       // N
  float* tot   = ncnt + NN;                       // 1 (padded to 4)
  unsigned* mxkey = (unsigned*)(tot + 4);         // N*4
  // fragment-major bf16 packed weights (16B aligned)
  unsigned short* pWin = (unsigned short*)(mxkey + (size_t)NN * NHEAD); // 384*128
  unsigned short* pWl  = pWin + (size_t)DINF * HIDF;                    // 128*512
  unsigned short* pWr  = pWl + (size_t)HIDF * NHEAD * HIDF;             // 128*512
  unsigned short* pWe  = pWr + (size_t)HIDF * NHEAD * HIDF;             // 384*512

  // global init
  k_zero<<<4, 256, 0, stream>>>(esum, DINF);
  k_zero<<<256, 256, 0, stream>>>(avg, EEDG);
  k_zero<<<128, 256, 0, stream>>>(nsum, NN);
  k_zero<<<128, 256, 0, stream>>>(ncnt, NN);
  k_zero<<<1, 32, 0, stream>>>(tot, 1);

  k_colmean<<<(EEDG + 255) / 256, 128, 0, stream>>>(ea, esum);
  k_qb<<<1, HIDF, 0, stream>>>(qe, Wq, bq, bin, qb);

  // h0 = x @ W_in + (b_in + q@Wq + bq)
  k_packB<<<64, 256, 0, stream>>>(Win, pWin, DINF, HIDF);
  k_gemm_bias<<<dim3((NN + 63) / 64, 1), 256, 0, stream>>>(x, pWin, qb, h, NN, DINF, HIDF);

  for (int i = 0; i < NLAY; ++i) {
    const float* Wli = Wl + (size_t)i * HIDF * NHEAD * HIDF;
    const float* Wri = Wr + (size_t)i * HIDF * NHEAD * HIDF;
    const float* Wei = We + (size_t)i * DINF * NHEAD * HIDF;
    const float* bli = bl + (size_t)i * NHEAD * HIDF;
    const float* bri = br + (size_t)i * NHEAD * HIDF;
    const float* atti = att + (size_t)i * NHEAD * HIDF;
    const float* gbi = gbias + (size_t)i * HIDF;
    const float* gi  = lng + (size_t)i * HIDF;
    const float* bi  = lnb + (size_t)i * HIDF;

    k_packB<<<64, 256, 0, stream>>>(Wli, pWl, HIDF, NHEAD * HIDF);
    k_packB<<<64, 256, 0, stream>>>(Wri, pWr, HIDF, NHEAD * HIDF);
    k_packB<<<192, 256, 0, stream>>>(Wei, pWe, DINF, NHEAD * HIDF);
    k_eWm<<<2, 256, 0, stream>>>(esum, Wei, eWm);
    k_zero<<<32, 256, 0, stream>>>(den, NN * NHEAD);
    k_zero<<<512, 256, 0, stream>>>(acc, NN * HIDF);
    k_zero<<<32, 256, 0, stream>>>((float*)mxkey, NN * NHEAD); // bits 0 == -inf key

    k_gemm_bias<<<dim3((NN + 63) / 64, 4), 256, 0, stream>>>(h, pWl, bli, xl, NN, HIDF, NHEAD * HIDF);
    k_gemm_bias<<<dim3((NN + 63) / 64, 4), 256, 0, stream>>>(h, pWr, bri, xr, NN, HIDF, NHEAD * HIDF);

    k_edge_logits<<<EEDG / 32, 256, 0, stream>>>(ea, pWe, xl, xr, atti, src, dst, lgt, mxkey);
    k_self_logits<<<(NN * NHEAD + 255) / 256, 256, 0, stream>>>(xl, xr, eWm, atti, lgt, mxkey);
    k_expsum<<<(EATOT * NHEAD + 255) / 256, 256, 0, stream>>>(lgt, mxkey, den, dst);
    k_aggregate<<<(EATOT + 7) / 8, 256, 0, stream>>>(lgt, den, xl, src, dst, acc, avg);
    k_finalize<<<(NN + 7) / 8, 256, 0, stream>>>(acc, h, gbi, gi, bi, (i < NLAY - 1) ? 1 : 0);
  }

  k_natt_accum<<<(EEDG + 255) / 256, 256, 0, stream>>>(avg, dst, nsum, ncnt);
  k_natt_div<<<(NN + 255) / 256, 256, 0, stream>>>(nsum, ncnt, tot);
  k_natt_out<<<(NN + 255) / 256, 256, 0, stream>>>(nsum, tot, out + (size_t)NN * HIDF);
  k_copy<<<1024, 256, 0, stream>>>(h, out, NN * HIDF);
}
